// RNNModel_36043365548225
// MI455X (gfx1250) — matrix operand order repaired and checked by reading, not compiled, not run on hardware
//
#include <hip/hip_runtime.h>
#include <hip/hip_bf16.h>

typedef __bf16 bf16_t;
typedef __attribute__((ext_vector_type(16))) __bf16 v16bf;
typedef __attribute__((ext_vector_type(8)))  __bf16 v8bf;
typedef __attribute__((ext_vector_type(8)))  float  v8f;

#define SEQ   2000
#define BATCH 64
#define NI    3
#define NN    512      // observable state width
#define NPI   515      // N + NI (K dimension of the GEMM)
#define KT    17       // K tiles of 32 -> 544 padded
#define KPAD  (KT * 32)
#define ITILES 32      // 512 / 16 output tiles in i
#define EFF_DT 0.05f

// ---------------------------------------------------------------------------
// Pre-pack M (f32, row-major 515x515) into the WMMA B-operand layout, bf16.
// Bpack[it][kt][lane][e]:  lane%16 = output column offset i within the 16-wide i tile,
// K offset j = (lane/16)*8 + (e&7) + (e>>3)*16 within the 32-wide K tile.
// B[j][i] = M[i][j]; out-of-range K (j >= 515) zero-padded.
// ---------------------------------------------------------------------------
__global__ void prepack_M(const float* __restrict__ M, bf16_t* __restrict__ Bpack) {
    int idx = blockIdx.x * blockDim.x + threadIdx.x;
    const int total = ITILES * KT * 32 * 16;
    if (idx >= total) return;
    int e    = idx & 15;
    int lane = (idx >> 4) & 31;
    int kt   = (idx >> 9) % KT;
    int it   = (idx >> 9) / KT;
    int i = it * 16 + (lane & 15);          // output index (row of M)
    int j = kt * 32 + (lane >> 4) * 8 + (e & 7) + (e >> 3) * 16;       // K index (col of M)
    float v = (j < NPI) ? M[(size_t)i * NPI + j] : 0.0f;
    Bpack[idx] = (bf16_t)v;
}

// ---------------------------------------------------------------------------
// Persistent single-workgroup RNN scan. 512 threads = 16 waves on one WGP.
// Wave w owns i-tiles {2w, 2w+1} and all 4 batch-tiles (8 accumulators).
// x state is kept in registers in C-matrix layout across all 2000 steps.
// ---------------------------------------------------------------------------
__global__ __launch_bounds__(512, 1) void rnn_scan(
    const float* __restrict__ r_in,   // (SEQ, B, NI) f32
    const float* __restrict__ h,      // (NPI) f32
    const bf16_t* __restrict__ Bpack, // packed weights
    float* __restrict__ out)          // outputs (SEQ,B,N) then Xs (SEQ,B,N)
{
    __shared__ bf16_t comb[BATCH * KPAD];   // activations, bf16, K-padded

    const int tid  = threadIdx.x;
    const int lane = tid & 31;
    const int w    = tid >> 5;            // wave id 0..15 -> i-tile pair
    const int rA   = lane & 15;           // row-in-tile (A) / col-in-tile (C)
    const int hi   = lane >> 4;           // upper-half-lane flag

    // --- init combined = [rate(0) x 512, r_in[0], zeros-pad] --------------
    const float r0 = 1.0f / (1.0f + __expf(1.0f));   // sigmoid(2*(0-0.5))
    for (int idx = tid; idx < BATCH * KPAD; idx += (int)blockDim.x) {
        int b = idx / KPAD;
        int j = idx - b * KPAD;
        float v;
        if (j < NN)        v = r0;
        else if (j < NPI)  v = r_in[b * NI + (j - NN)];
        else               v = 0.0f;
        comb[idx] = (bf16_t)v;
    }

    // per-lane bias for the two owned i-tiles (i = itile*16 + (lane&15))
    float hv[2];
    hv[0] = h[(2 * w + 0) * 16 + rA];
    hv[1] = h[(2 * w + 1) * 16 + rA];

    // x state registers, C-layout: x[bt][it2][v] = x[b = bt*16+v+hi*8][i]
    v8f x[4][2];
    #pragma unroll
    for (int bt = 0; bt < 4; ++bt)
        #pragma unroll
        for (int it2 = 0; it2 < 2; ++it2)
            x[bt][it2] = (v8f){0.f, 0.f, 0.f, 0.f, 0.f, 0.f, 0.f, 0.f};

    const size_t XS_OFF = (size_t)SEQ * BATCH * NN;

    __syncthreads();

    #pragma unroll 1
    for (int t = 0; t < SEQ; ++t) {
        // ---------------- GEMM phase: acc = combined @ M.T ----------------
        v8f acc[4][2];
        #pragma unroll
        for (int bt = 0; bt < 4; ++bt)
            #pragma unroll
            for (int it2 = 0; it2 < 2; ++it2)
                acc[bt][it2] = (v8f){0.f, 0.f, 0.f, 0.f, 0.f, 0.f, 0.f, 0.f};

        #pragma unroll 1
        for (int kt = 0; kt < KT; ++kt) {
            // A fragments: lane<16 holds K j0..j0+7 (e0-7) and j0+16..j0+23
            // (e8-15); lane>=16 is shifted by +8, per 16-bit A layout.
            const int j0 = kt * 32 + hi * 8;
            v16bf a[4];
            #pragma unroll
            for (int bt = 0; bt < 4; ++bt) {
                const bf16_t* p = &comb[(bt * 16 + rA) * KPAD + j0];
                v8bf lo = *(const v8bf*)p;
                v8bf hb = *(const v8bf*)(p + 16);
                a[bt] = __builtin_shufflevector(lo, hb,
                        0,1,2,3,4,5,6,7,8,9,10,11,12,13,14,15);
            }
            #pragma unroll
            for (int it2 = 0; it2 < 2; ++it2) {
                const bf16_t* q = Bpack +
                    ((((size_t)(2 * w + it2)) * KT + kt) * 32 + lane) * 16;
                v8bf lo = *(const v8bf*)q;
                v8bf hb = *(const v8bf*)(q + 8);
                v16bf bfrag = __builtin_shufflevector(lo, hb,
                        0,1,2,3,4,5,6,7,8,9,10,11,12,13,14,15);
                #pragma unroll
                for (int bt = 0; bt < 4; ++bt) {
                    acc[bt][it2] = __builtin_amdgcn_wmma_f32_16x16x32_bf16(
                        false, a[bt], false, bfrag,
                        (short)0, acc[bt][it2], false, false);
                }
            }
        }

        __syncthreads();   // all reads of comb for step t complete

        // ---------------- update phase: x, r, outputs ---------------------
        #pragma unroll
        for (int it2 = 0; it2 < 2; ++it2) {
            const int i = (2 * w + it2) * 16 + rA;
            #pragma unroll
            for (int bt = 0; bt < 4; ++bt) {
                #pragma unroll
                for (int v = 0; v < 8; ++v) {
                    const int b = bt * 16 + v + hi * 8;
                    float xo = x[bt][it2][v];
                    float xn = xo + (-xo + acc[bt][it2][v] + hv[it2]) * EFF_DT;
                    x[bt][it2][v] = xn;
                    // r = F0 * sigmoid(BETA0*(x - THETA0)), F0=1,BETA0=2,THETA0=0.5
                    float r = 1.0f / (1.0f + __expf(-2.0f * (xn - 0.5f)));
                    size_t o = (size_t)t * (BATCH * NN) + (size_t)b * NN + i;
                    out[o]          = r;    // outputs
                    out[XS_OFF + o] = xn;   // Xs
                    comb[b * KPAD + i] = (bf16_t)r;  // next step's activations
                }
            }
        }
        // stage r_in[t+1] into the tail of combined
        if ((t + 1 < SEQ) && (tid < BATCH * NI)) {
            comb[(tid / NI) * KPAD + NN + (tid % NI)] =
                (bf16_t)r_in[(size_t)(t + 1) * (BATCH * NI) + tid];
        }
        __syncthreads();   // comb ready for step t+1
    }
}

// ---------------------------------------------------------------------------
extern "C" void kernel_launch(void* const* d_in, const int* in_sizes, int n_in,
                              void* d_out, int out_size, void* d_ws, size_t ws_size,
                              hipStream_t stream) {
    const float* r_in = (const float*)d_in[0];   // (2000, 64, 3)
    const float* M    = (const float*)d_in[1];   // (515, 515)
    const float* h    = (const float*)d_in[2];   // (515,)
    (void)in_sizes; (void)n_in; (void)out_size; (void)ws_size;

    bf16_t* Bpack = (bf16_t*)d_ws;               // 32*17*32*16*2B = 544 KB

    const int total = ITILES * KT * 32 * 16;
    prepack_M<<<(total + 255) / 256, 256, 0, stream>>>(M, Bpack);
    rnn_scan<<<1, 512, 0, stream>>>(r_in, h, Bpack, (float*)d_out);
}
